// PocMoE_12945031430544
// MI455X (gfx1250) — matrix operand order repaired and checked by reading, not compiled, not run on hardware
//
#include <hip/hip_runtime.h>
#include <hip/hip_bf16.h>

typedef __attribute__((ext_vector_type(16))) __bf16 v16bf;
typedef __attribute__((ext_vector_type(8)))  float  v8f;

union FragAB { uint4 q[2]; v16bf v; };

__device__ __forceinline__ unsigned short f2bf_rne(float f) {
  unsigned int u = __float_as_uint(f);
  unsigned int r = u + 0x7FFFu + ((u >> 16) & 1u);
  return (unsigned short)(r >> 16);
}

// Issue a per-lane 16B async copy global -> LDS (tracked by ASYNCcnt).
__device__ __forceinline__ void async_copy16(const unsigned short* g, unsigned int lds_off) {
  asm volatile("global_load_async_to_lds_b128 %0, %1, off"
               :: "v"(lds_off), "v"(g) : "memory");
}

// ---------------------------------------------------------------------------
// Router: one block per token; wave w computes expert w's logit; thread 0 does
// the 8-wide softmax. logits -> d_out slice, weights -> workspace.
// ---------------------------------------------------------------------------
__global__ __launch_bounds__(256)
void moe_router(const float* __restrict__ x, const float* __restrict__ gw,
                const float* __restrict__ gb, float* __restrict__ logits,
                float* __restrict__ rw, int D, int E) {
  const int t = blockIdx.x;
  const int tid = threadIdx.x;
  const int e = tid >> 5;        // 8 waves == 8 experts
  const int lane = tid & 31;
  const float* xr = x + (size_t)t * D;
  float p = 0.0f;
  for (int d = lane; d < D; d += 32)
    p += xr[d] * gw[(size_t)d * E + e];
  #pragma unroll
  for (int off = 16; off > 0; off >>= 1)
    p += __shfl_down(p, off, 32);
  __shared__ float sl[8];
  if (lane == 0) sl[e] = p + gb[e];
  __syncthreads();
  if (tid == 0) {
    float mx = sl[0];
    #pragma unroll
    for (int i = 1; i < 8; ++i) mx = fmaxf(mx, sl[i]);
    float ex[8]; float s = 0.0f;
    #pragma unroll
    for (int i = 0; i < 8; ++i) { ex[i] = __expf(sl[i] - mx); s += ex[i]; }
    const float inv = 1.0f / s;
    #pragma unroll
    for (int i = 0; i < 8; ++i) {
      logits[(size_t)t * 8 + i] = sl[i];
      rw[(size_t)t * 8 + i] = ex[i] * inv;
    }
  }
}

// ---------------------------------------------------------------------------
// Elementwise fp32 -> bf16 (x): 4 elems/thread.
// ---------------------------------------------------------------------------
__global__ __launch_bounds__(256)
void convert_bf16(const float* __restrict__ src, unsigned short* __restrict__ dst,
                  long long n4) {
  const long long i = (long long)blockIdx.x * blockDim.x + threadIdx.x;
  if (i < n4) {
    const float4 f = ((const float4*)src)[i];
    uint2 o;
    o.x = (unsigned)f2bf_rne(f.x) | ((unsigned)f2bf_rne(f.y) << 16);
    o.y = (unsigned)f2bf_rne(f.z) | ((unsigned)f2bf_rne(f.w) << 16);
    ((uint2*)dst)[i] = o;
  }
}

// ---------------------------------------------------------------------------
// Per-expert transpose+convert: src [E,K,N] fp32 -> dst [E,N,K] bf16.
// 32x32 tiles through LDS.
// ---------------------------------------------------------------------------
__global__ __launch_bounds__(256)
void transpose_to_bf16(const float* __restrict__ src, unsigned short* __restrict__ dst,
                       int K, int N) {
  __shared__ float t[32][33];
  const size_t eo = (size_t)blockIdx.z * K * N;
  const int k0 = blockIdx.y * 32, n0 = blockIdx.x * 32;
  const int r = threadIdx.x >> 5, c = threadIdx.x & 31;
  #pragma unroll
  for (int j = 0; j < 4; ++j)
    t[r + j * 8][c] = src[eo + (size_t)(k0 + r + j * 8) * N + n0 + c];
  __syncthreads();
  #pragma unroll
  for (int j = 0; j < 4; ++j)
    dst[eo + (size_t)(n0 + r + j * 8) * K + k0 + c] = f2bf_rne(t[c][r + j * 8]);
}

// ---------------------------------------------------------------------------
// WMMA GEMM: C[M,N] = A[M,K] @ Bt[N,K]^T (+bias, +epilogue). A,Bt bf16.
// Block = 256 threads (8 waves), macro-tile 128x128, k-step 32, double-buffered
// LDS filled by global_load_async_to_lds_b128 (ASYNCcnt pipeline).
// Wave (wm,wn): wm in {0,1}, wn in {0..3} owns 64x32 -> 4x2 wmma subtiles.
// MODE 0: epilogue relu(+bias) -> bf16 h
// MODE 1: epilogue +bias -> expert_out, fused weighted accumulation into final
// ---------------------------------------------------------------------------
template <int MODE>
__global__ __launch_bounds__(256)
void moe_gemm(const unsigned short* __restrict__ A,   // [M,K] bf16
              const unsigned short* __restrict__ Bt,  // [N,K] bf16
              const float* __restrict__ bias,
              unsigned short* __restrict__ hOut,      // MODE 0
              const float* __restrict__ rw,           // MODE 1
              float* __restrict__ eOut,               // MODE 1
              float* __restrict__ fOut,               // MODE 1
              int expert, int M, int N, int K, int firstExpert) {
  __shared__ __align__(16) unsigned short lA[2][128 * 32];  // [row][k]
  __shared__ __align__(16) unsigned short lB[2][128 * 32];  // [n][k]

  const int tid   = threadIdx.x;
  const int lane  = tid & 31;
  const int wid   = tid >> 5;
  const int waveM = wid >> 2;
  const int waveN = wid & 3;
  const int half  = lane >> 4;
  const int l16   = lane & 15;

  const int n0 = blockIdx.x * 128;
  const int m0 = blockIdx.y * 128;

  // per-thread staging coordinates: 512 16B slots per tile, 2 per thread per matrix
  const int srow = tid >> 2;        // 0..63 (and +64 on second slot)
  const int sk   = (tid & 3) << 3;  // k offset in elements (0,8,16,24)

  v8f acc[4][2];
  #pragma unroll
  for (int mt = 0; mt < 4; ++mt)
    #pragma unroll
    for (int nt = 0; nt < 2; ++nt)
      #pragma unroll
      for (int i = 0; i < 8; ++i) acc[mt][nt][i] = 0.0f;

  const int KSTEPS = K >> 5;

  auto stage = [&](int buf, int ks) {
    const int k0 = ks << 5;
    #pragma unroll
    for (int j = 0; j < 2; ++j) {
      const int row = srow + j * 64;
      async_copy16(A  + (size_t)(m0 + row) * K + k0 + sk,
                   (unsigned int)(uintptr_t)&lA[buf][row * 32 + sk]);
      async_copy16(Bt + (size_t)(n0 + row) * K + k0 + sk,
                   (unsigned int)(uintptr_t)&lB[buf][row * 32 + sk]);
    }
  };

  stage(0, 0);
  int buf = 0;
  for (int ks = 0; ks < KSTEPS; ++ks) {
    if (ks + 1 < KSTEPS) {
      stage(buf ^ 1, ks + 1);
      // 8 outstanding (4 current + 4 next); retire current tile's 4 (in-order)
      asm volatile("s_wait_asynccnt 0x4" ::: "memory");
    } else {
      asm volatile("s_wait_asynccnt 0x0" ::: "memory");
    }
    __syncthreads();

    // ---- compute: 8 WMMA per wave per k-step ----
    FragAB fb[2];
    #pragma unroll
    for (int nt = 0; nt < 2; ++nt) {
      const int col = waveN * 32 + nt * 16 + l16;
      const uint4* pb = (const uint4*)&lB[buf][col * 32];
      fb[nt].q[0] = pb[half];
      fb[nt].q[1] = pb[2 + half];
    }
    #pragma unroll
    for (int mt = 0; mt < 4; ++mt) {
      const int row = waveM * 64 + mt * 16 + l16;
      const uint4* pa = (const uint4*)&lA[buf][row * 32];
      FragAB fa;
      fa.q[0] = pa[half];           // K 0-7 / 8-15
      fa.q[1] = pa[2 + half];       // K 16-23 / 24-31
      #pragma unroll
      for (int nt = 0; nt < 2; ++nt) {
        acc[mt][nt] = __builtin_amdgcn_wmma_f32_16x16x32_bf16(
            false, fa.v, false, fb[nt].v, (short)0, acc[mt][nt], false, false);
      }
    }
    __syncthreads();
    buf ^= 1;
  }

  // ---- epilogue (C/D layout: col = lane%16, row = r + 8*(lane/16)) ----
  #pragma unroll
  for (int mt = 0; mt < 4; ++mt) {
    #pragma unroll
    for (int nt = 0; nt < 2; ++nt) {
      const int col = n0 + waveN * 32 + nt * 16 + l16;
      const float bv = bias[col];
      #pragma unroll
      for (int r = 0; r < 8; ++r) {
        const int row = m0 + waveM * 64 + mt * 16 + half * 8 + r;
        float v = acc[mt][nt][r] + bv;
        if (MODE == 0) {
          v = fmaxf(v, 0.0f);
          hOut[(size_t)row * N + col] = f2bf_rne(v);
        } else {
          eOut[((size_t)row * 8 + expert) * N + col] = v;
          const float wv = rw[(size_t)row * 8 + expert] * v;
          const size_t fo = (size_t)row * N + col;
          if (firstExpert) fOut[fo] = wv;
          else             fOut[fo] += wv;
        }
      }
    }
  }
}

// ---------------------------------------------------------------------------
extern "C" void kernel_launch(void* const* d_in, const int* in_sizes, int n_in,
                              void* d_out, int out_size, void* d_ws, size_t ws_size,
                              hipStream_t stream) {
  (void)in_sizes; (void)n_in; (void)out_size; (void)ws_size;
  const float* x      = (const float*)d_in[0];   // [B,S,D]
  const float* gate_w = (const float*)d_in[1];   // [D,E]
  const float* gate_b = (const float*)d_in[2];   // [E]
  const float* w1     = (const float*)d_in[3];   // [E,D,F]
  const float* b1     = (const float*)d_in[4];   // [E,F]
  const float* w2     = (const float*)d_in[5];   // [E,F,D]
  const float* b2     = (const float*)d_in[6];   // [E,D]

  const int Bb = 4, S = 2048, D = 1024, E = 8, F = 2048;
  const int M = Bb * S;                          // 8192 tokens

  float* out_final  = (float*)d_out;                       // [M,D]
  float* out_logits = out_final + (size_t)M * D;           // [M,E]
  float* out_expert = out_logits + (size_t)M * E;          // [M,E,D]

  // workspace layout
  float*          rw   = (float*)d_ws;                             // [M,E]
  unsigned short* xb   = (unsigned short*)(rw + (size_t)M * E);    // [M,D]  bf16
  unsigned short* w1t  = xb  + (size_t)M * D;                      // [E,F,D] bf16 (transposed)
  unsigned short* w2t  = w1t + (size_t)E * F * D;                  // [E,D,F] bf16 (transposed)
  unsigned short* hbuf = w2t + (size_t)E * D * F;                  // [M,F]  bf16

  moe_router<<<M, 256, 0, stream>>>(x, gate_w, gate_b, out_logits, rw, D, E);

  // one-time precision/layout conversion (bandwidth-bound, runs once)
  {
    const long long n4 = (long long)M * D / 4;
    convert_bf16<<<(unsigned)((n4 + 255) / 256), 256, 0, stream>>>(x, xb, n4);
    transpose_to_bf16<<<dim3(F / 32, D / 32, E), 256, 0, stream>>>(w1, w1t, D, F);
    transpose_to_bf16<<<dim3(D / 32, F / 32, E), 256, 0, stream>>>(w2, w2t, F, D);
  }

  for (int e = 0; e < E; ++e) {
    // h = relu(x @ w1[e] + b1[e])   [M,F], stored bf16
    moe_gemm<0><<<dim3(F / 128, M / 128), 256, 0, stream>>>(
        xb, w1t + (size_t)e * F * D, b1 + (size_t)e * F,
        hbuf, nullptr, nullptr, nullptr, e, M, F, D, 0);
    // out_e = h @ w2[e] + b2[e]     [M,D]; fused weighted accumulation
    moe_gemm<1><<<dim3(D / 128, M / 128), 256, 0, stream>>>(
        hbuf, w2t + (size_t)e * D * F, b2 + (size_t)e * D,
        nullptr, rw, out_expert, out_final, e, M, D, F, (e == 0) ? 1 : 0);
  }
}
